// MegaLayer_78692390798024
// MI455X (gfx1250) — compile-verified, matrix operand order checked by reading, not run on hardware
//
#include <hip/hip_runtime.h>
#include <hip/hip_bf16.h>
#include <math.h>

// ---------------- types for WMMA ----------------
typedef _Float16 v16h __attribute__((ext_vector_type(16)));
typedef _Float16 h8   __attribute__((ext_vector_type(8)));
typedef float    v8f  __attribute__((ext_vector_type(8)));

union V16 { v16h v; h8 h[2]; };

#define BATCH 2
#define SEQ 4096
#define DIM 128
#define HEADS 16
#define DQK 64
#define DV 256
#define CHUNK 128
#define NCHUNK 32   // SEQ / CHUNK
#define PROJ_ROWS 8
#define FIN_ROWS 8

__device__ __forceinline__ float sigmoidf_(float x) { return 1.0f / (1.0f + __expf(-x)); }
__device__ __forceinline__ float siluf_(float x)    { return x * sigmoidf_(x); }

// ---------------- relative position bias table ----------------
// bias[rel] = rel_table[bucket(rel)] * sqrt(DQK), rel in [0, SEQ)
__global__ void bias_kernel(const float* __restrict__ table, float* __restrict__ bias) {
    int rel = blockIdx.x * blockDim.x + threadIdx.x;
    if (rel >= SEQ) return;
    int bucket;
    if (rel < 16) {
        bucket = rel;
    } else {
        int v = 16 + (int)(logf((float)rel / 16.0f) / logf(8.0f) * 16.0f);
        bucket = v < 31 ? v : 31;
    }
    bias[rel] = table[bucket] * 8.0f;   // sqrt(64) = 8
}

// ---------------- EMA: chunked parallel scan ----------------
__global__ void ema_passA(const float* __restrict__ x,
                          const float* __restrict__ expansion,
                          const float* __restrict__ reduction,
                          const float* __restrict__ alphas,
                          const float* __restrict__ dampen,
                          float* __restrict__ ema_local,
                          float* __restrict__ E) {
    const int b = blockIdx.y;
    const int c = blockIdx.x;
    const int d = threadIdx.x;          // 0..127
    float q[HEADS], ex[HEADS], rd[HEADS], s[HEADS];
#pragma unroll
    for (int h = 0; h < HEADS; h++) {
        float av = sigmoidf_(alphas[h]);
        float dv = sigmoidf_(dampen[h]);
        q[h]  = (1.0f - av) * dv;
        ex[h] = av * expansion[h * DIM + d];
        rd[h] = reduction[h * DIM + d];
        s[h]  = 0.0f;
    }
    const size_t base = ((size_t)b * SEQ + (size_t)c * CHUNK) * DIM + d;
    const float* xp = x + base;
    float* op = ema_local + base;
    for (int t = 0; t < CHUNK; t++) {
        float xv = xp[(size_t)t * DIM];
        float acc = 0.0f;
#pragma unroll
        for (int h = 0; h < HEADS; h++) {
            s[h] = q[h] * s[h] + ex[h] * xv;
            acc += s[h] * rd[h];
        }
        op[(size_t)t * DIM] = acc;
    }
#pragma unroll
    for (int h = 0; h < HEADS; h++)
        E[(((size_t)b * NCHUNK + c) * HEADS + h) * DIM + d] = s[h];
}

__global__ void ema_passB(const float* __restrict__ alphas,
                          const float* __restrict__ dampen,
                          const float* __restrict__ E,
                          float* __restrict__ Cc) {
    int idx = blockIdx.x * blockDim.x + threadIdx.x;   // 0..4095
    if (idx >= BATCH * HEADS * DIM) return;
    int b = idx >> 11;
    int h = (idx >> 7) & (HEADS - 1);
    int d = idx & (DIM - 1);
    float av = sigmoidf_(alphas[h]);
    float q  = (1.0f - av) * sigmoidf_(dampen[h]);
    float qL = powf(q, (float)CHUNK);
    float S = 0.0f;
    for (int c = 0; c < NCHUNK; c++) {
        size_t o = (((size_t)b * NCHUNK + c) * HEADS + h) * DIM + d;
        Cc[o] = S;
        S = E[o] + qL * S;
    }
}

__global__ void ema_passC(const float* __restrict__ alphas,
                          const float* __restrict__ dampen,
                          const float* __restrict__ reduction,
                          const float* __restrict__ Cc,
                          float* __restrict__ ema) {
    const int b = blockIdx.y;
    const int c = blockIdx.x;
    const int d = threadIdx.x;
    float q[HEADS], p[HEADS], ch[HEADS];
#pragma unroll
    for (int h = 0; h < HEADS; h++) {
        float av = sigmoidf_(alphas[h]);
        q[h] = (1.0f - av) * sigmoidf_(dampen[h]);
        p[h] = q[h];
        ch[h] = reduction[h * DIM + d] *
                Cc[(((size_t)b * NCHUNK + c) * HEADS + h) * DIM + d];
    }
    float* op = ema + ((size_t)b * SEQ + (size_t)c * CHUNK) * DIM + d;
    for (int t = 0; t < CHUNK; t++) {
        float acc = op[(size_t)t * DIM];
#pragma unroll
        for (int h = 0; h < HEADS; h++) {
            acc += p[h] * ch[h];
            p[h] *= q[h];
        }
        op[(size_t)t * DIM] = acc;
    }
}

// ---------------- fused projections (8 rows per block, weight reuse) ----------
__global__ __launch_bounds__(256)
void proj_kernel(const float* __restrict__ ema, const float* __restrict__ x,
                 const float* __restrict__ Wqk, const float* __restrict__ bqk,
                 const float* __restrict__ gamma, const float* __restrict__ beta,
                 const float* __restrict__ Wv, const float* __restrict__ bv,
                 const float* __restrict__ Wr, const float* __restrict__ br,
                 const float* __restrict__ Wu, const float* __restrict__ bu,
                 const float* __restrict__ Wh, const float* __restrict__ bh,
                 _Float16* __restrict__ qh, _Float16* __restrict__ kh,
                 _Float16* __restrict__ vt, float* __restrict__ reset,
                 float* __restrict__ upd, float* __restrict__ hpre) {
    const int row0 = blockIdx.x * PROJ_ROWS;   // 8 consecutive rows
    const int tid = threadIdx.x;               // 0..255
    __shared__ float se[PROJ_ROWS][DIM];
    __shared__ float sx[PROJ_ROWS][DIM];
    for (int idx = tid; idx < PROJ_ROWS * DIM; idx += 256) {
        int r = idx >> 7, k = idx & (DIM - 1);
        se[r][k] = ema[(size_t)(row0 + r) * DIM + k];
        sx[r][k] = x[(size_t)(row0 + r) * DIM + k];
    }
    __syncthreads();

    float acc[PROJ_ROWS];
    for (int c = tid; c < DQK + DV + DV + DIM + DIM; c += 256) {
        if (c < DQK) {                       // qk -> q,k
#pragma unroll
            for (int r = 0; r < PROJ_ROWS; r++) acc[r] = bqk[c];
            for (int k = 0; k < DIM; k++) {
                float w = Wqk[k * DQK + c];
#pragma unroll
                for (int r = 0; r < PROJ_ROWS; r++) acc[r] += se[r][k] * w;
            }
#pragma unroll
            for (int r = 0; r < PROJ_ROWS; r++) {
                float s = siluf_(acc[r]);
                size_t row = row0 + r;
                qh[row * DQK + c] = (_Float16)(s * gamma[c]       + beta[c]);
                kh[row * DQK + c] = (_Float16)(s * gamma[DQK + c] + beta[DQK + c]);
            }
        } else if (c < DQK + DV) {           // v (transposed store)
            int c2 = c - DQK;
#pragma unroll
            for (int r = 0; r < PROJ_ROWS; r++) acc[r] = bv[c2];
            for (int k = 0; k < DIM; k++) {
                float w = Wv[k * DV + c2];
#pragma unroll
                for (int r = 0; r < PROJ_ROWS; r++) acc[r] += sx[r][k] * w;
            }
#pragma unroll
            for (int r = 0; r < PROJ_ROWS; r++) {
                int row = row0 + r;
                int b = row >> 12, n = row & (SEQ - 1);
                vt[((size_t)b * DV + c2) * SEQ + n] = (_Float16)siluf_(acc[r]);
            }
        } else if (c < DQK + 2 * DV) {       // reset gate
            int c2 = c - DQK - DV;
#pragma unroll
            for (int r = 0; r < PROJ_ROWS; r++) acc[r] = br[c2];
            for (int k = 0; k < DIM; k++) {
                float w = Wr[k * DV + c2];
#pragma unroll
                for (int r = 0; r < PROJ_ROWS; r++) acc[r] += se[r][k] * w;
            }
#pragma unroll
            for (int r = 0; r < PROJ_ROWS; r++)
                reset[(size_t)(row0 + r) * DV + c2] = siluf_(acc[r]);
        } else if (c < DQK + 2 * DV + DIM) { // update gate
            int c2 = c - DQK - 2 * DV;
#pragma unroll
            for (int r = 0; r < PROJ_ROWS; r++) acc[r] = bu[c2];
            for (int k = 0; k < DIM; k++) {
                float w = Wu[k * DIM + c2];
#pragma unroll
                for (int r = 0; r < PROJ_ROWS; r++) acc[r] += se[r][k] * w;
            }
#pragma unroll
            for (int r = 0; r < PROJ_ROWS; r++)
                upd[(size_t)(row0 + r) * DIM + c2] = sigmoidf_(acc[r]);
        } else {                             // ema@Wh + bh (pre-activation)
            int c2 = c - DQK - 2 * DV - DIM;
#pragma unroll
            for (int r = 0; r < PROJ_ROWS; r++) acc[r] = bh[c2];
            for (int k = 0; k < DIM; k++) {
                float w = Wh[k * DIM + c2];
#pragma unroll
                for (int r = 0; r < PROJ_ROWS; r++) acc[r] += se[r][k] * w;
            }
#pragma unroll
            for (int r = 0; r < PROJ_ROWS; r++)
                hpre[(size_t)(row0 + r) * DIM + c2] = acc[r];
        }
    }
}

// ---------------- flash-style laplacian attention (WMMA f16) ----------------
// Block: 256 threads = 8 waves, 32 query rows (2 x 16-row tiles), DV=256 cols.
// Wave w (qi = w>>2, kw = w&3): computes the S tile (q-tile qi, key sub-tile
// kb0+16*kw), writes laplacian(P) to LDS; then owns 32 V-columns and runs
// P @ V for BOTH q-tiles (each V B-tile loaded once, used by two WMMAs).
// Bias table staged in LDS; per-step bias values hoisted into registers
// BEFORE the QK WMMAs so their LDS latency hides under the matrix ops.
__global__ __launch_bounds__(256)
void attn_kernel(const _Float16* __restrict__ qh,
                 const _Float16* __restrict__ kh,
                 const _Float16* __restrict__ vt,
                 const float* __restrict__ bias,
                 float* __restrict__ attn) {
    const int b   = blockIdx.y;
    const int qt  = blockIdx.x;             // 32-row query tile
    const int tid = threadIdx.x;            // 0..255
    const int wv  = tid >> 5;               // wave 0..7
    const int qi  = wv >> 2;                // q sub-tile this wave scores
    const int kw  = wv & 3;                 // key sub-tile this wave scores
    const int lane  = tid & 31;
    const int lhalf = lane & 15;
    const int hi    = lane >> 4;            // 0/1
    const int fb    = hi * 8;               // K-offset base for A/B layout

    __shared__ __align__(16) _Float16 plds[2 * 16 * 64]; // P: 2 q-tiles x 16 x 64 keys
    __shared__ float blds[SEQ];                          // staged bias table (16 KB)

    const _Float16* Q = qh + (size_t)b * SEQ * DQK;
    const _Float16* K = kh + (size_t)b * SEQ * DQK;
    const _Float16* V = vt + (size_t)b * DV * SEQ;
    float* O = attn + (size_t)b * SEQ * DV;

    const int qmax = qt * 32 + 31;
    for (int i = tid; i <= qmax; i += 256) blds[i] = bias[i];

    // this wave's Q sub-tile as A-matrix (16x64 = two 16x32 A tiles)
    const int qrow = qt * 32 + qi * 16 + lhalf;
    V16 aq0, aq1;
    aq0.h[0] = *(const h8*)(Q + (size_t)qrow * DQK + fb);
    aq0.h[1] = *(const h8*)(Q + (size_t)qrow * DQK + fb + 16);
    aq1.h[0] = *(const h8*)(Q + (size_t)qrow * DQK + 32 + fb);
    aq1.h[1] = *(const h8*)(Q + (size_t)qrow * DQK + 32 + fb + 16);

    v8f zero = {};
    v8f acc[4];                    // [q-tile 0/1][n-tile 0/1]
#pragma unroll
    for (int i = 0; i < 4; i++) acc[i] = zero;

    const float scale = 1.0f / (float)SEQ;
    const float mu  = 0.70710678f;          // sqrt(0.5)
    const float inv = 0.79788456f;          // 1/(std*sqrt(2)) = 1/sqrt(pi/2)

    __syncthreads();               // blds ready

    for (int kb0 = 0; kb0 <= qmax; kb0 += 64) {
        const int kt = kb0 + kw * 16;       // this wave's key sub-tile
        const int krow = kt + lhalf;

        // hoisted bias prefetch: entries rel0..rel0+7 (contiguous in r),
        // independent of the WMMA result -> LDS latency hides under QK WMMAs
        const int rel0 = (qt * 32 + qi * 16 + hi * 8) - (kt + lhalf);
        float bvv[8];
#pragma unroll
        for (int r = 0; r < 8; r++) {
            const int rel = rel0 + r;
            const int relc = rel > 0 ? rel : 0;
            bvv[r] = blds[relc];
        }

        V16 bk0, bk1;
        bk0.h[0] = *(const h8*)(K + (size_t)krow * DQK + fb);
        bk0.h[1] = *(const h8*)(K + (size_t)krow * DQK + fb + 16);
        bk1.h[0] = *(const h8*)(K + (size_t)krow * DQK + 32 + fb);
        bk1.h[1] = *(const h8*)(K + (size_t)krow * DQK + 32 + fb + 16);

        v8f s = zero;
        s = __builtin_amdgcn_wmma_f32_16x16x32_f16(false, aq0.v, false, bk0.v,
                                                   (short)0, s, false, false);
        s = __builtin_amdgcn_wmma_f32_16x16x32_f16(false, aq1.v, false, bk1.v,
                                                   (short)0, s, false, false);
        // scale + bias + laplacian + causal mask -> LDS P (branchless)
#pragma unroll
        for (int r = 0; r < 8; r++) {
            const int m = r + hi * 8;
            const int rel = rel0 + r;
            float sv = s[r] * scale + bvv[r];
            float p = 0.5f * (1.0f + erff((sv - mu) * inv));
            p = (rel >= 0) ? p : 0.0f;
            plds[qi * 1024 + m * 64 + kw * 16 + lhalf] = (_Float16)p;
        }
        __syncthreads();
        // P @ V for this wave's 32 columns, both q-tiles
#pragma unroll
        for (int kk = 0; kk < 2; kk++) {
            V16 ap0, ap1;
            ap0.h[0] = *(const h8*)(plds + lhalf * 64 + kk * 32 + fb);
            ap0.h[1] = *(const h8*)(plds + lhalf * 64 + kk * 32 + fb + 16);
            ap1.h[0] = *(const h8*)(plds + 1024 + lhalf * 64 + kk * 32 + fb);
            ap1.h[1] = *(const h8*)(plds + 1024 + lhalf * 64 + kk * 32 + fb + 16);
#pragma unroll
            for (int nt = 0; nt < 2; nt++) {
                const int col = wv * 32 + nt * 16 + lhalf;
                V16 bv_;
                bv_.h[0] = *(const h8*)(V + (size_t)col * SEQ + kb0 + kk * 32 + fb);
                bv_.h[1] = *(const h8*)(V + (size_t)col * SEQ + kb0 + kk * 32 + fb + 16);
                acc[nt]     = __builtin_amdgcn_wmma_f32_16x16x32_f16(false, ap0.v, false, bv_.v,
                                                                     (short)0, acc[nt], false, false);
                acc[2 + nt] = __builtin_amdgcn_wmma_f32_16x16x32_f16(false, ap1.v, false, bv_.v,
                                                                     (short)0, acc[2 + nt], false, false);
            }
        }
        __syncthreads();
    }
    // store accumulators (C layout)
#pragma unroll
    for (int qj = 0; qj < 2; qj++) {
#pragma unroll
        for (int nt = 0; nt < 2; nt++) {
#pragma unroll
            for (int r = 0; r < 8; r++) {
                const int m = r + hi * 8;
                const int row = qt * 32 + qj * 16 + m;
                const int col = wv * 32 + nt * 16 + lhalf;
                O[(size_t)row * DV + col] = acc[qj * 2 + nt][r];
            }
        }
    }
}

// ---------------- final gating + output (8 rows per block) ----------------
// gated = attn_out * reset;  H = silu(hpre + gated @ Uh);  out = u*H + (1-u)*x
__global__ __launch_bounds__(128)
void final_kernel(const float* __restrict__ attn,
                  const float* __restrict__ reset,
                  const float* __restrict__ upd,
                  const float* __restrict__ hpre,
                  const float* __restrict__ Uh,
                  const float* __restrict__ x,
                  float* __restrict__ out) {
    const int row0 = blockIdx.x * FIN_ROWS;
    const int d = threadIdx.x;          // 0..127
    __shared__ float g[FIN_ROWS][DV];
    for (int idx = d; idx < FIN_ROWS * DV; idx += 128) {
        int r = idx >> 8, k = idx & (DV - 1);
        size_t o = (size_t)(row0 + r) * DV + k;
        g[r][k] = attn[o] * reset[o];
    }
    __syncthreads();
    float acc[FIN_ROWS];
#pragma unroll
    for (int r = 0; r < FIN_ROWS; r++) acc[r] = hpre[(size_t)(row0 + r) * DIM + d];
    for (int k = 0; k < DV; k++) {
        float u = Uh[k * DIM + d];
#pragma unroll
        for (int r = 0; r < FIN_ROWS; r++) acc[r] += g[r][k] * u;
    }
#pragma unroll
    for (int r = 0; r < FIN_ROWS; r++) {
        size_t o = (size_t)(row0 + r) * DIM + d;
        float H = siluf_(acc[r]);
        float u = upd[o];
        out[o] = u * H + (1.0f - u) * x[o];
    }
}

// ---------------- launch ----------------
extern "C" void kernel_launch(void* const* d_in, const int* in_sizes, int n_in,
                              void* d_out, int out_size, void* d_ws, size_t ws_size,
                              hipStream_t stream) {
    const float* x         = (const float*)d_in[0];
    const float* expansion = (const float*)d_in[1];
    const float* reduction = (const float*)d_in[2];
    const float* alphas    = (const float*)d_in[3];
    const float* dampen    = (const float*)d_in[4];
    const float* Wqk       = (const float*)d_in[5];
    const float* bqk       = (const float*)d_in[6];
    const float* gamma     = (const float*)d_in[7];
    const float* beta      = (const float*)d_in[8];
    const float* Wv        = (const float*)d_in[9];
    const float* bv        = (const float*)d_in[10];
    const float* rel_table = (const float*)d_in[11];
    const float* Wr        = (const float*)d_in[12];
    const float* br        = (const float*)d_in[13];
    const float* Wu        = (const float*)d_in[14];
    const float* bu        = (const float*)d_in[15];
    const float* Wh        = (const float*)d_in[16];
    const float* Uh        = (const float*)d_in[17];
    const float* bh        = (const float*)d_in[18];
    float* out = (float*)d_out;

    // workspace carve-up (all offsets 256B-aligned)
    char* w = (char*)d_ws;
    size_t off = 0;
    float*    ema   = (float*)(w + off);    off += (size_t)BATCH * SEQ * DIM * 4;
    float*    E     = (float*)(w + off);    off += (size_t)BATCH * NCHUNK * HEADS * DIM * 4;
    float*    Cc    = (float*)(w + off);    off += (size_t)BATCH * NCHUNK * HEADS * DIM * 4;
    _Float16* qh    = (_Float16*)(w + off); off += (size_t)BATCH * SEQ * DQK * 2;
    _Float16* kh    = (_Float16*)(w + off); off += (size_t)BATCH * SEQ * DQK * 2;
    _Float16* vt    = (_Float16*)(w + off); off += (size_t)BATCH * DV * SEQ * 2;
    float*    reset = (float*)(w + off);    off += (size_t)BATCH * SEQ * DV * 4;
    float*    upd   = (float*)(w + off);    off += (size_t)BATCH * SEQ * DIM * 4;
    float*    hpre  = (float*)(w + off);    off += (size_t)BATCH * SEQ * DIM * 4;
    float*    attn  = (float*)(w + off);    off += (size_t)BATCH * SEQ * DV * 4;
    float*    bias  = (float*)(w + off);    off += (size_t)SEQ * 4;
    (void)ws_size; (void)n_in; (void)in_sizes; (void)out_size;

    bias_kernel<<<SEQ / 256, 256, 0, stream>>>(rel_table, bias);

    ema_passA<<<dim3(NCHUNK, BATCH), DIM, 0, stream>>>(x, expansion, reduction,
                                                       alphas, dampen, ema, E);
    ema_passB<<<(BATCH * HEADS * DIM) / 128, 128, 0, stream>>>(alphas, dampen, E, Cc);
    ema_passC<<<dim3(NCHUNK, BATCH), DIM, 0, stream>>>(alphas, dampen, reduction, Cc, ema);

    proj_kernel<<<BATCH * SEQ / PROJ_ROWS, 256, 0, stream>>>(ema, x, Wqk, bqk, gamma, beta,
                                                             Wv, bv, Wr, br, Wu, bu, Wh, bh,
                                                             qh, kh, vt, reset, upd, hpre);

    attn_kernel<<<dim3(SEQ / 32, BATCH), 256, 0, stream>>>(qh, kh, vt, bias, attn);

    final_kernel<<<BATCH * SEQ / FIN_ROWS, 128, 0, stream>>>(attn, reset, upd, hpre, Uh, x, out);
}
